// PromptSelector_43817256353966
// MI455X (gfx1250) — matrix operand order repaired and checked by reading, not compiled, not run on hardware
//
#include <hip/hip_runtime.h>

// ---------------------------------------------------------------------------
// MI455X (gfx1250, wave32) implementation.
// Key insight: outputs depend only on x[:,0,:] -> only K/V GEMMs are "big".
// Those run on v_wmma_f32_16x16x32_f16 with a software-pipelined (ping-pong)
// K-loop so loads of tile i+1 overlap the WMMAs of tile i. Everything touching
// the 8 cls rows stays fp32 VALU for numeric fidelity of the top-k ordering.
// ---------------------------------------------------------------------------

typedef __attribute__((ext_vector_type(16))) _Float16 v16h;
typedef __attribute__((ext_vector_type(8)))  float    v8f;

#define D    768
#define H    16
#define DK   48
#define DFF  3072
#define POOL 100
#define NCLS 7
#define SQ   1024        // padded sequence (1 cls + 1023 text)
#define BSZ  8
#define NTOK (BSZ * SQ)  // 8192

// ---------------- block reductions (fixed order -> deterministic) ----------
__device__ __forceinline__ float block_reduce_sum(float v, float* sh) {
    int t = threadIdx.x;
    sh[t] = v; __syncthreads();
    for (int s = blockDim.x >> 1; s > 0; s >>= 1) {
        if (t < s) sh[t] += sh[t + s];
        __syncthreads();
    }
    float r = sh[0]; __syncthreads();
    return r;
}
__device__ __forceinline__ float block_reduce_max(float v, float* sh) {
    int t = threadIdx.x;
    sh[t] = v; __syncthreads();
    for (int s = blockDim.x >> 1; s > 0; s >>= 1) {
        if (t < s) sh[t] = fmaxf(sh[t], sh[t + s]);
        __syncthreads();
    }
    float r = sh[0]; __syncthreads();
    return r;
}

// ---------------- K1: build x row, LayerNorm1 -> h16 (f16), cls rows f32 ---
__global__ __launch_bounds__(256) void k_ln1(const float* __restrict__ text,
                                             const float* __restrict__ cls_tok,
                                             const float* __restrict__ ga,
                                             const float* __restrict__ gb,
                                             _Float16* __restrict__ h16,
                                             float* __restrict__ h0f32) {
    __shared__ float red[256];
    int tok = blockIdx.x;
    int b = tok >> 10, s = tok & (SQ - 1);
    const float* src = (s == 0) ? cls_tok
                                : text + (size_t)(b * (SQ - 1) + (s - 1)) * D;
    int t = threadIdx.x;
    float v0 = src[t], v1 = src[t + 256], v2 = src[t + 512];
    float mean = block_reduce_sum(v0 + v1 + v2, red) * (1.0f / (float)D);
    float d0 = v0 - mean, d1 = v1 - mean, d2 = v2 - mean;
    float var = block_reduce_sum(d0 * d0 + d1 * d1 + d2 * d2, red) * (1.0f / (float)(D - 1));
    float inv = 1.0f / (sqrtf(var) + 1e-6f);
    float y0 = ga[t]       * d0 * inv + gb[t];
    float y1 = ga[t + 256] * d1 * inv + gb[t + 256];
    float y2 = ga[t + 512] * d2 * inv + gb[t + 512];
    _Float16* hr = h16 + (size_t)tok * D;
    hr[t] = (_Float16)y0; hr[t + 256] = (_Float16)y1; hr[t + 512] = (_Float16)y2;
    if (s == 0) {
        float* cr = h0f32 + (size_t)b * D;
        cr[t] = y0; cr[t + 256] = y1; cr[t + 512] = y2;
    }
}

// ---------------- K2: W (KxN f32) -> W^T (NxK f16) -------------------------
__global__ __launch_bounds__(256) void k_wt16(const float* __restrict__ W,
                                              _Float16* __restrict__ WT) {
    int idx = blockIdx.x * 256 + threadIdx.x;   // idx = n*D + k
    int n = idx / D, k = idx - n * D;
    WT[idx] = (_Float16)W[(size_t)k * D + n];
}

// ---------------- K3: C[M,N] = A[M,K]f16 @ B(K,N) + bias, via WMMA ---------
// BT is B transposed: [N][K] f16.  One wave computes a 16x64 tile.
// Software pipelined: ping-pong fragment buffers so the 10 b128 loads of the
// next k-tile are in flight while the 4 WMMAs of the current tile execute.
struct Frag { union { v16h v; uint4 u[2]; }; };

__device__ __forceinline__ void load_a(Frag& f, const _Float16* Arow, int kk, int half) {
    f.u[0] = *(const uint4*)(Arow + kk + 8 * half);
    f.u[1] = *(const uint4*)(Arow + kk + 16 + 8 * half);
}
__device__ __forceinline__ void load_b(Frag* f, const _Float16* B0, int kk, int half) {
    const int K = D;
    int bo = kk + 8 * half;
    f[0].u[0] = *(const uint4*)(B0 + bo);          f[0].u[1] = *(const uint4*)(B0 + bo + 16);
    f[1].u[0] = *(const uint4*)(B0 + 16 * K + bo); f[1].u[1] = *(const uint4*)(B0 + 16 * K + bo + 16);
    f[2].u[0] = *(const uint4*)(B0 + 32 * K + bo); f[2].u[1] = *(const uint4*)(B0 + 32 * K + bo + 16);
    f[3].u[0] = *(const uint4*)(B0 + 48 * K + bo); f[3].u[1] = *(const uint4*)(B0 + 48 * K + bo + 16);
}

__global__ __launch_bounds__(32) void k_gemm_h16(const _Float16* __restrict__ A,
                                                 const _Float16* __restrict__ BT,
                                                 const float* __restrict__ bias,
                                                 float* __restrict__ C) {
    const int K = D, N = D;
    int m0 = blockIdx.x * 16;
    int n0 = blockIdx.y * 64;
    int lane = threadIdx.x;
    int half = lane >> 4, l15 = lane & 15;

    const _Float16* Arow = A + (size_t)(m0 + l15) * K;
    const _Float16* B0   = BT + (size_t)(n0 + l15) * K;

    v8f c0 = {}, c1 = {}, c2 = {}, c3 = {};
    Frag aP, aQ, bP[4], bQ[4];

    load_a(aP, Arow, 0, half);
    load_b(bP, B0,   0, half);

    #pragma unroll 1
    for (int kk = 0; kk < K; kk += 64) {
        // prefetch odd tile while even tile computes
        load_a(aQ, Arow, kk + 32, half);
        load_b(bQ, B0,   kk + 32, half);
        c0 = __builtin_amdgcn_wmma_f32_16x16x32_f16(false, aP.v, false, bP[0].v, (short)0, c0, false, false);
        c1 = __builtin_amdgcn_wmma_f32_16x16x32_f16(false, aP.v, false, bP[1].v, (short)0, c1, false, false);
        c2 = __builtin_amdgcn_wmma_f32_16x16x32_f16(false, aP.v, false, bP[2].v, (short)0, c2, false, false);
        c3 = __builtin_amdgcn_wmma_f32_16x16x32_f16(false, aP.v, false, bP[3].v, (short)0, c3, false, false);
        // prefetch next even tile while odd tile computes
        if (kk + 64 < K) {
            load_a(aP, Arow, kk + 64, half);
            load_b(bP, B0,   kk + 64, half);
        }
        c0 = __builtin_amdgcn_wmma_f32_16x16x32_f16(false, aQ.v, false, bQ[0].v, (short)0, c0, false, false);
        c1 = __builtin_amdgcn_wmma_f32_16x16x32_f16(false, aQ.v, false, bQ[1].v, (short)0, c1, false, false);
        c2 = __builtin_amdgcn_wmma_f32_16x16x32_f16(false, aQ.v, false, bQ[2].v, (short)0, c2, false, false);
        c3 = __builtin_amdgcn_wmma_f32_16x16x32_f16(false, aQ.v, false, bQ[3].v, (short)0, c3, false, false);
    }

    for (int r = 0; r < 8; ++r) {
        int row = m0 + r + 8 * half;
        float* Crow = C + (size_t)row * N;
        int c = n0 + l15;
        Crow[c]      = c0[r] + bias[c];
        Crow[c + 16] = c1[r] + bias[c + 16];
        Crow[c + 32] = c2[r] + bias[c + 32];
        Crow[c + 48] = c3[r] + bias[c + 48];
    }
}

// ---------------- K4: attention for the 8 cls queries (fp32) ---------------
// one block per (batch, head); 256 threads
__global__ __launch_bounds__(256) void k_attn(const float* __restrict__ h0,
                                              const float* __restrict__ Wq,
                                              const float* __restrict__ bq,
                                              const float* __restrict__ Km,
                                              const float* __restrict__ Vm,
                                              float* __restrict__ o8) {
    __shared__ float xs[D];
    __shared__ float sc[SQ];
    __shared__ float qv[DK];
    __shared__ float part[DK][4];
    __shared__ float red[256];
    int bh = blockIdx.x;
    int b = bh >> 4, h = bh & 15;
    int t = threadIdx.x;

    xs[t] = h0[b * D + t]; xs[t + 256] = h0[b * D + t + 256]; xs[t + 512] = h0[b * D + t + 512];
    __syncthreads();

    // q = ln1(x0) @ Wq + bq   (48 outputs, 4-way split-K per output)
    if (t < 192) {
        int dk = t >> 2, st = t & 3;
        float acc = 0.0f;
        int k0 = st * 192;
        for (int k = k0; k < k0 + 192; ++k)
            acc += xs[k] * Wq[(size_t)k * D + h * DK + dk];
        part[dk][st] = acc;
    }
    __syncthreads();
    if (t < DK) qv[t] = part[t][0] + part[t][1] + part[t][2] + part[t][3] + bq[h * DK + t];
    __syncthreads();

    // scores over 1024 keys
    const float scale = 0.14433756729740643f;   // 1/sqrt(48)
    float lmax = -1e30f;
    for (int j = t; j < SQ; j += 256) {
        const float* kr = Km + (size_t)(b * SQ + j) * D + h * DK;
        float acc = 0.0f;
        for (int dk = 0; dk < DK; ++dk) acc += qv[dk] * kr[dk];
        acc *= scale;
        sc[j] = acc;
        lmax = fmaxf(lmax, acc);
    }
    float mx = block_reduce_max(lmax, red);
    float lsum = 0.0f;
    for (int j = t; j < SQ; j += 256) { float e = expf(sc[j] - mx); sc[j] = e; lsum += e; }
    float inv = 1.0f / block_reduce_sum(lsum, red);
    for (int j = t; j < SQ; j += 256) sc[j] *= inv;
    __syncthreads();

    // o = p @ V   (48 outputs, 4-way split over keys)
    if (t < 192) {
        int dk = t >> 2, st = t & 3;
        float acc = 0.0f;
        int j0 = st * 256;
        for (int j = j0; j < j0 + 256; ++j)
            acc += sc[j] * Vm[(size_t)(b * SQ + j) * D + h * DK + dk];
        part[dk][st] = acc;
    }
    __syncthreads();
    if (t < DK) o8[b * D + h * DK + t] = part[t][0] + part[t][1] + part[t][2] + part[t][3];
}

// ---------------- K5a: out-proj + residual + LN2 (8 rows) ------------------
__global__ __launch_bounds__(256) void k_proj_ln2(const float* __restrict__ o8,
                                                  const float* __restrict__ Wo,
                                                  const float* __restrict__ bo,
                                                  const float* __restrict__ cls_tok,
                                                  const float* __restrict__ ga,
                                                  const float* __restrict__ gb,
                                                  float* __restrict__ x1,
                                                  float* __restrict__ h2) {
    __shared__ float osh[D];
    __shared__ float red[256];
    int b = blockIdx.x, t = threadIdx.x;
    osh[t] = o8[b * D + t]; osh[t + 256] = o8[b * D + t + 256]; osh[t + 512] = o8[b * D + t + 512];
    __syncthreads();
    float vals[3];
    for (int i = 0; i < 3; ++i) {
        int n = t + i * 256;
        float acc = 0.0f;
        for (int k = 0; k < D; ++k) acc += osh[k] * Wo[(size_t)k * D + n];
        float xv = cls_tok[n] + acc + bo[n];
        vals[i] = xv;
        x1[b * D + n] = xv;
    }
    __syncthreads();
    float mean = block_reduce_sum(vals[0] + vals[1] + vals[2], red) * (1.0f / (float)D);
    float d0 = vals[0] - mean, d1 = vals[1] - mean, d2 = vals[2] - mean;
    float var = block_reduce_sum(d0 * d0 + d1 * d1 + d2 * d2, red) * (1.0f / (float)(D - 1));
    float inv = 1.0f / (sqrtf(var) + 1e-6f);
    h2[b * D + t]       = ga[t]       * d0 * inv + gb[t];
    h2[b * D + t + 256] = ga[t + 256] * d1 * inv + gb[t + 256];
    h2[b * D + t + 512] = ga[t + 512] * d2 * inv + gb[t + 512];
}

// ---------------- K5b: FFN layer 1 (8 rows) --------------------------------
__global__ __launch_bounds__(256) void k_ffn1(const float* __restrict__ h2,
                                              const float* __restrict__ W1,
                                              const float* __restrict__ b1,
                                              float* __restrict__ hff) {
    __shared__ float hs[D];
    int b = blockIdx.x, t = threadIdx.x;
    hs[t] = h2[b * D + t]; hs[t + 256] = h2[b * D + t + 256]; hs[t + 512] = h2[b * D + t + 512];
    __syncthreads();
    for (int i = 0; i < 12; ++i) {
        int n = t + i * 256;
        float acc = 0.0f;
        for (int k = 0; k < D; ++k) acc += hs[k] * W1[(size_t)k * DFF + n];
        hff[b * DFF + n] = fmaxf(acc + b1[n], 0.0f);
    }
}

// ---------------- K5c: FFN layer 2 + residual + final LN (8 rows) ----------
__global__ __launch_bounds__(256) void k_ffn2_lnf(const float* __restrict__ hff,
                                                  const float* __restrict__ W2,
                                                  const float* __restrict__ b2,
                                                  const float* __restrict__ x1,
                                                  const float* __restrict__ ga,
                                                  const float* __restrict__ gb,
                                                  float* __restrict__ cls) {
    __shared__ float hs[DFF];
    __shared__ float red[256];
    int b = blockIdx.x, t = threadIdx.x;
    for (int i = 0; i < 12; ++i) hs[t + i * 256] = hff[b * DFF + t + i * 256];
    __syncthreads();
    float vals[3];
    for (int i = 0; i < 3; ++i) {
        int n = t + i * 256;
        float acc = 0.0f;
        for (int k = 0; k < DFF; ++k) acc += hs[k] * W2[(size_t)k * D + n];
        vals[i] = acc + b2[n] + x1[b * D + n];
    }
    float mean = block_reduce_sum(vals[0] + vals[1] + vals[2], red) * (1.0f / (float)D);
    float d0 = vals[0] - mean, d1 = vals[1] - mean, d2 = vals[2] - mean;
    float var = block_reduce_sum(d0 * d0 + d1 * d1 + d2 * d2, red) * (1.0f / (float)(D - 1));
    float inv = 1.0f / (sqrtf(var) + 1e-6f);
    cls[b * D + t]       = ga[t]       * d0 * inv + gb[t];
    cls[b * D + t + 256] = ga[t + 256] * d1 * inv + gb[t + 256];
    cls[b * D + t + 512] = ga[t + 512] * d2 * inv + gb[t + 512];
}

// ---------------- K5d: heads, entropy->k, top-k ordering -------------------
__global__ __launch_bounds__(128) void k_head(const float* __restrict__ cls,
                                              const float* __restrict__ Wpool,
                                              const float* __restrict__ bpool,
                                              const float* __restrict__ Wcls,
                                              const float* __restrict__ bcls,
                                              float* __restrict__ out_logits,
                                              int* __restrict__ top_idx,
                                              int* __restrict__ adjk) {
    __shared__ float cs[D];
    __shared__ float pl[POOL];
    __shared__ float cl[NCLS];
    int b = blockIdx.x, t = threadIdx.x;
    for (int i = t; i < D; i += 128) cs[i] = cls[b * D + i];
    __syncthreads();
    if (t < POOL) {
        float acc = 0.0f;
        for (int k = 0; k < D; ++k) acc += cs[k] * Wpool[(size_t)k * POOL + t];
        pl[t] = acc + bpool[t];
    }
    if (t < NCLS) {
        float acc = 0.0f;
        for (int k = 0; k < D; ++k) acc += cs[k] * Wcls[(size_t)k * NCLS + t];
        float v = acc + bcls[t];
        cl[t] = v;
        out_logits[b * NCLS + t] = v;
    }
    __syncthreads();
    if (t == 0) {
        // entropy of softmax(class_logits) -> adjusted_k
        float mx = cl[0];
        for (int i = 1; i < NCLS; ++i) mx = fmaxf(mx, cl[i]);
        float e[NCLS], s = 0.0f;
        for (int i = 0; i < NCLS; ++i) { e[i] = expf(cl[i] - mx); s += e[i]; }
        float ent = 0.0f;
        for (int i = 0; i < NCLS; ++i) { float p = e[i] / s; ent -= p * logf(p + 1e-9f); }
        float norm = ent / logf(7.0f);
        int k = (int)floorf(1.0f + norm * (float)(POOL - 1));
        k = k < 1 ? 1 : (k > POOL ? POOL : k);
        adjk[b] = k;
        // top-k by pool logits (monotonic with softmax probs), descending,
        // earliest index wins ties (strict >)
        float v[POOL]; int id[POOL];
        for (int i = 0; i < POOL; ++i) { v[i] = pl[i]; id[i] = i; }
        for (int i = 0; i < POOL; ++i) {
            int best = i;
            for (int j = i + 1; j < POOL; ++j) if (v[j] > v[best]) best = j;
            float tv = v[i]; v[i] = v[best]; v[best] = tv;
            int ti = id[i]; id[i] = id[best]; id[best] = ti;
            top_idx[b * POOL + i] = id[i];
        }
    }
}

// ---------------- K5e: gather/mask selected prompts ------------------------
__global__ __launch_bounds__(256) void k_prompts(const float* __restrict__ pool,
                                                 const int* __restrict__ top_idx,
                                                 const int* __restrict__ adjk,
                                                 float* __restrict__ out) {
    int blk = blockIdx.x;            // blk = b*POOL + j
    int b = blk / POOL, j = blk - b * POOL;
    int t = threadIdx.x;
    bool keep = j < adjk[b];
    int row = top_idx[blk];
    const float* src = pool + (size_t)row * D;
    float* dst = out + (size_t)blk * D;
    dst[t]       = keep ? src[t]       : 0.0f;
    dst[t + 256] = keep ? src[t + 256] : 0.0f;
    dst[t + 512] = keep ? src[t + 512] : 0.0f;
}

// ---------------------------------------------------------------------------
extern "C" void kernel_launch(void* const* d_in, const int* in_sizes, int n_in,
                              void* d_out, int out_size, void* d_ws, size_t ws_size,
                              hipStream_t stream) {
    const float* text    = (const float*)d_in[0];
    const float* pool    = (const float*)d_in[1];
    const float* cls_tok = (const float*)d_in[2];
    const float* Wq = (const float*)d_in[3];   const float* bq = (const float*)d_in[4];
    const float* Wk = (const float*)d_in[5];   const float* bk = (const float*)d_in[6];
    const float* Wv = (const float*)d_in[7];   const float* bv = (const float*)d_in[8];
    const float* Wo = (const float*)d_in[9];   const float* bo = (const float*)d_in[10];
    const float* l1a = (const float*)d_in[11]; const float* l1b = (const float*)d_in[12];
    const float* l2a = (const float*)d_in[13]; const float* l2b = (const float*)d_in[14];
    const float* W1 = (const float*)d_in[15];  const float* b1 = (const float*)d_in[16];
    const float* W2 = (const float*)d_in[17];  const float* b2 = (const float*)d_in[18];
    const float* lfa = (const float*)d_in[19]; const float* lfb = (const float*)d_in[20];
    const float* Wpool = (const float*)d_in[21]; const float* bpool = (const float*)d_in[22];
    const float* Wcls  = (const float*)d_in[23]; const float* bcls  = (const float*)d_in[24];

    char* ws = (char*)d_ws;
    size_t off = 0;
    auto alloc = [&](size_t bytes) { void* p = ws + off; off += (bytes + 255) & ~(size_t)255; return p; };
    _Float16* h16   = (_Float16*)alloc((size_t)NTOK * D * 2);
    _Float16* WkT16 = (_Float16*)alloc((size_t)D * D * 2);
    _Float16* WvT16 = (_Float16*)alloc((size_t)D * D * 2);
    float*    Kf32  = (float*)   alloc((size_t)NTOK * D * 4);
    float*    Vf32  = (float*)   alloc((size_t)NTOK * D * 4);
    float*    h0f32 = (float*)   alloc((size_t)BSZ * D * 4);
    float*    o8    = (float*)   alloc((size_t)BSZ * D * 4);
    float*    x1    = (float*)   alloc((size_t)BSZ * D * 4);
    float*    h2    = (float*)   alloc((size_t)BSZ * D * 4);
    float*    hff   = (float*)   alloc((size_t)BSZ * DFF * 4);
    float*    clsb  = (float*)   alloc((size_t)BSZ * D * 4);
    int*      tidx  = (int*)     alloc((size_t)BSZ * POOL * 4);
    int*      adjk  = (int*)     alloc((size_t)BSZ * 4);

    float* out_prompts = (float*)d_out;                          // [8,100,768]
    float* out_logits  = (float*)d_out + (size_t)BSZ * POOL * D; // [8,7]

    k_ln1 <<<NTOK, 256, 0, stream>>>(text, cls_tok, l1a, l1b, h16, h0f32);
    k_wt16<<<(D * D) / 256, 256, 0, stream>>>(Wk, WkT16);
    k_wt16<<<(D * D) / 256, 256, 0, stream>>>(Wv, WvT16);
    k_gemm_h16<<<dim3(NTOK / 16, D / 64), 32, 0, stream>>>(h16, WkT16, bk, Kf32);
    k_gemm_h16<<<dim3(NTOK / 16, D / 64), 32, 0, stream>>>(h16, WvT16, bv, Vf32);
    k_attn<<<BSZ * H, 256, 0, stream>>>(h0f32, Wq, bq, Kf32, Vf32, o8);
    k_proj_ln2<<<BSZ, 256, 0, stream>>>(o8, Wo, bo, cls_tok, l2a, l2b, x1, h2);
    k_ffn1<<<BSZ, 256, 0, stream>>>(h2, W1, b1, hff);
    k_ffn2_lnf<<<BSZ, 256, 0, stream>>>(hff, W2, b2, x1, lfa, lfb, clsb);
    k_head<<<BSZ, 128, 0, stream>>>(clsb, Wpool, bpool, Wcls, bcls, out_logits, tidx, adjk);
    k_prompts<<<BSZ * POOL, 256, 0, stream>>>(pool, tidx, adjk, out_prompts);
}
